// MultiHeadSelfAttention_21569325760708
// MI455X (gfx1250) — matrix operand order repaired and checked by reading, not compiled, not run on hardware
//
#include <hip/hip_runtime.h>
#include <cstdint>
#include <cstddef>

// ---------------------------------------------------------------------------
// MHA forward for MI455X (gfx1250, wave32, WMMA).
// All GEMMs in bf16 with fp32 accumulation via v_wmma_f32_16x16x32_bf16.
// ---------------------------------------------------------------------------

#define DIM_   512
#define NH_    8
#define HD_    64
#define B_     8
#define L_     1024
#define NTOK_  (B_ * L_)          // 8192
#define N_QKV_ (3 * DIM_)         // 1536

typedef __bf16 bf16_t;
typedef __attribute__((ext_vector_type(16))) __bf16 v16bf;
typedef __attribute__((ext_vector_type(8)))  __bf16 v8bf;
typedef __attribute__((ext_vector_type(8)))  float  v8f;

union V16 { v16bf v; v8bf h[2]; };

static __device__ __forceinline__ v8f wmma_bf16f32(v16bf a, v16bf b, v8f c) {
  // D = A(16x32 bf16) * B(32x16 bf16) + C(16x16 f32)
  return __builtin_amdgcn_wmma_f32_16x16x32_bf16(
      /*neg_a=*/false, a, /*neg_b=*/false, b,
      /*c_mod=*/(short)0, c, /*reuse_a=*/false, /*reuse_b=*/false);
}

// ---------------------------------------------------------------------------
// fp32 -> bf16 conversion
// ---------------------------------------------------------------------------
__global__ void f32_to_bf16_kernel(const float* __restrict__ in,
                                   bf16_t* __restrict__ out, int n) {
  int i = blockIdx.x * blockDim.x + threadIdx.x;
  if (i < n) out[i] = (bf16_t)in[i];
}

// ---------------------------------------------------------------------------
// Shared WMMA GEMM mainloop: block tile 256(M) x 64(N), K-step 32.
// 256 threads = 8 waves; wave w owns rows [32w, 32w+32) (two 16-row
// sub-tiles u=0,1), 4 column tiles. 8 WMMAs per wave per k-step with only
// 12 ds_load_b128 (each B fragment is reused by both row sub-tiles).
// A staged row-major in LDS (stride 40 bf16), B staged transposed [n][k]
// (stride 48 bf16) so B fragment halves are contiguous 16B chunks.
// ---------------------------------------------------------------------------
static __device__ __forceinline__ void gemm_tile(
    const bf16_t* __restrict__ A, int lda,
    const bf16_t* __restrict__ Bm, int ldb,
    int kdim, int rowBase, int colBase,
    bf16_t* As, bf16_t* Bs, v8f acc[2][4]) {
  const int tid = threadIdx.x;
  const int w = tid >> 5;
  const int lane = tid & 31;
  const int lane16 = lane & 15;
  const int khalf = lane >> 4;

  const v8f zacc = {0.f, 0.f, 0.f, 0.f, 0.f, 0.f, 0.f, 0.f};
#pragma unroll
  for (int u = 0; u < 2; ++u)
#pragma unroll
    for (int t = 0; t < 4; ++t) acc[u][t] = zacc;

  for (int k0 = 0; k0 < kdim; k0 += 32) {
    __syncthreads();
    {  // A tile: 256 x 32; each thread moves one full 64B row chunk
      const v8bf* src = (const v8bf*)(A + (size_t)(rowBase + tid) * lda + k0);
      v8bf x0 = src[0], x1 = src[1], x2 = src[2], x3 = src[3];
      v8bf* dst = (v8bf*)&As[tid * 40];
      dst[0] = x0; dst[1] = x1; dst[2] = x2; dst[3] = x3;
    }
    {  // B tile: 32(k) x 64(n) from row-major, stored transposed Bs[n][k]
      int k = tid & 31, ngrp = tid >> 5;
      v8bf bv = *(const v8bf*)(Bm + (size_t)(k0 + k) * ldb + colBase + ngrp * 8);
#pragma unroll
      for (int j = 0; j < 8; ++j) Bs[(ngrp * 8 + j) * 48 + k] = bv[j];
    }
    __syncthreads();

    // A fragments: lane m = lane16, element e -> k = (e>>3)*16 + khalf*8 + (e&7)
    V16 af[2];
#pragma unroll
    for (int u = 0; u < 2; ++u) {
      const int row = w * 32 + u * 16 + lane16;
      af[u].h[0] = *(const v8bf*)&As[row * 40 + khalf * 8];
      af[u].h[1] = *(const v8bf*)&As[row * 40 + 16 + khalf * 8];
    }
#pragma unroll
    for (int t = 0; t < 4; ++t) {
      // B fragment: lane n = lane16, element e -> k = (e>>3)*16 + khalf*8 + (e&7)
      V16 bf;
      bf.h[0] = *(const v8bf*)&Bs[(t * 16 + lane16) * 48 + khalf * 8];
      bf.h[1] = *(const v8bf*)&Bs[(t * 16 + lane16) * 48 + 16 + khalf * 8];
#pragma unroll
      for (int u = 0; u < 2; ++u)
        acc[u][t] = wmma_bf16f32(af[u].v, bf.v, acc[u][t]);
    }
  }
}

// ---------------------------------------------------------------------------
// QKV projection with fused RoPE epilogue.
//   A = xb [8192][512] bf16, B = wqkvb [512][1536] bf16.
//   grid = (24, 32). Each wave's 64 output cols are exactly one head, so the
//   rotate_half partner of col d lives in accumulator t^2 of the same lane.
//   Writes Q,K: [b][h][l][64] bf16 (Q pre-scaled by hd^-0.5),
//          Vt:  [b][h][64][l] bf16 (transposed for the PV GEMM).
// ---------------------------------------------------------------------------
__global__ void gemm_qkv_rope_kernel(const bf16_t* __restrict__ xb,
                                     const bf16_t* __restrict__ wqkvb,
                                     bf16_t* __restrict__ Qb,
                                     bf16_t* __restrict__ Kb,
                                     bf16_t* __restrict__ Vtb) {
  __shared__ bf16_t As[256 * 40];
  __shared__ bf16_t Bs[64 * 48];
  const int rowBase = blockIdx.y * 256;
  const int colBase = blockIdx.x * 64;
  v8f acc[2][4];
  gemm_tile(xb, DIM_, wqkvb, N_QKV_, DIM_, rowBase, colBase, As, Bs, acc);

  const int tid = threadIdx.x;
  const int w = tid >> 5, lane = tid & 31;
  const int lane16 = lane & 15, khalf = lane >> 4;
  const int sect = colBase / DIM_;           // 0=q 1=k 2=v
  const int h = (colBase % DIM_) / HD_;

#pragma unroll
  for (int u = 0; u < 2; ++u) {
    const int tokBase = rowBase + w * 32 + u * 16;
    const int b = tokBase / L_;
    const int lbase = tokBase % L_;
    if (sect < 2) {  // q or k -> RoPE
      bf16_t* dst = (sect == 0) ? Qb : Kb;
      const float qscale = (sect == 0) ? 0.125f : 1.0f;  // hd^-0.5 = 1/8
#pragma unroll
      for (int t = 0; t < 4; ++t) {
        const int d = t * 16 + lane16;
        const int dm = d & 31;
        const float invf = __powf(10000.0f, -(float)(2 * dm) * (1.0f / 64.0f));
        const float sign = (d < 32) ? -1.0f : 1.0f;
#pragma unroll
        for (int r = 0; r < 8; ++r) {
          const int l = lbase + r + khalf * 8;
          const float ang = (float)l * invf;
          float c, s;
          __sincosf(ang, &s, &c);
          const float val =
              (acc[u][t][r] * c + sign * acc[u][t ^ 2][r] * s) * qscale;
          dst[(((size_t)(b * NH_ + h) * L_ + l) * HD_) + d] = (bf16_t)val;
        }
      }
    } else {  // v -> transposed store
#pragma unroll
      for (int t = 0; t < 4; ++t) {
        const int d = t * 16 + lane16;
#pragma unroll
        for (int r = 0; r < 8; ++r) {
          const int l = lbase + r + khalf * 8;
          Vtb[(((size_t)(b * NH_ + h) * HD_ + d) * L_) + l] =
              (bf16_t)acc[u][t][r];
        }
      }
    }
  }
}

// ---------------------------------------------------------------------------
// Flash attention: grid = B*H*(L/64) = 1024 blocks of 128 threads (4 waves).
// Wave w handles a 16-query tile; online softmax over 64-key blocks.
// (16 query rows per wave keeps live VGPRs ~120; a 32-row tile would double
// oacc/sacc/stats and risk spills.)
// ---------------------------------------------------------------------------
__global__ void flash_attn_kernel(const bf16_t* __restrict__ Qb,
                                  const bf16_t* __restrict__ Kb,
                                  const bf16_t* __restrict__ Vtb,
                                  const unsigned char* __restrict__ mask,
                                  bf16_t* __restrict__ attnout) {
  __shared__ bf16_t Plds[4][16 * 72];  // per-wave P tile, row stride 72 bf16
  const int bid = blockIdx.x;
  const int qb = bid & 15;
  const int h = (bid >> 4) & 7;
  const int b = bid >> 7;
  const int w = threadIdx.x >> 5;
  const int lane = threadIdx.x & 31;
  const int lane16 = lane & 15, khalf = lane >> 4;

  const size_t bh = (size_t)(b * NH_ + h);
  const bf16_t* Qbh = Qb + bh * L_ * HD_;
  const bf16_t* Kbh = Kb + bh * L_ * HD_;
  const bf16_t* Vbh = Vtb + bh * HD_ * L_;
  const int qr0 = qb * 64 + w * 16;

  // Q fragments (A-layout), fixed across key blocks
  V16 qf[2];
#pragma unroll
  for (int s = 0; s < 2; ++s) {
    const bf16_t* qrow = Qbh + (size_t)(qr0 + lane16) * HD_ + s * 32 + khalf * 8;
    qf[s].h[0] = *(const v8bf*)qrow;
    qf[s].h[1] = *(const v8bf*)(qrow + 16);
  }

  const v8f zacc = {0.f, 0.f, 0.f, 0.f, 0.f, 0.f, 0.f, 0.f};
  float mrun[8], lsum[8];
  v8f oacc[4];
#pragma unroll
  for (int r = 0; r < 8; ++r) { mrun[r] = -1e30f; lsum[r] = 0.f; }
#pragma unroll
  for (int t = 0; t < 4; ++t) oacc[t] = zacc;

  bf16_t* Pw = &Plds[w][0];

  for (int kb = 0; kb < L_; kb += 64) {
    if (kb + 64 < L_)  // prefetch next key block
      __builtin_prefetch(Kbh + (size_t)(kb + 64) * HD_, 0, 0);

    // ---- S = (Q * hd^-0.5) @ K^T ----
    v8f sacc[4];
#pragma unroll
    for (int t = 0; t < 4; ++t) sacc[t] = zacc;
#pragma unroll
    for (int t = 0; t < 4; ++t) {
      const int key = kb + t * 16 + lane16;
      const bf16_t* krow = Kbh + (size_t)key * HD_;
#pragma unroll
      for (int s = 0; s < 2; ++s) {
        V16 kf;
        kf.h[0] = *(const v8bf*)(krow + s * 32 + khalf * 8);
        kf.h[1] = *(const v8bf*)(krow + s * 32 + khalf * 8 + 16);
        sacc[t] = wmma_bf16f32(qf[s].v, kf.v, sacc[t]);
      }
      if (!mask[b * L_ + key]) {  // every element of this lane has col `key`
#pragma unroll
        for (int r = 0; r < 8; ++r) sacc[t][r] += -1e9f;
      }
    }

    // ---- online softmax row stats (row m = r + khalf*8) ----
    float mnew[8], alpha[8];
#pragma unroll
    for (int r = 0; r < 8; ++r) {
      float tm = fmaxf(fmaxf(sacc[0][r], sacc[1][r]),
                       fmaxf(sacc[2][r], sacc[3][r]));
      for (int off = 1; off < 16; off <<= 1)
        tm = fmaxf(tm, __shfl_xor(tm, off, 32));
      mnew[r] = fmaxf(mrun[r], tm);
      alpha[r] = __expf(mrun[r] - mnew[r]);
      mrun[r] = mnew[r];
    }

    float psum[8];
#pragma unroll
    for (int r = 0; r < 8; ++r) psum[r] = 0.f;
#pragma unroll
    for (int t = 0; t < 4; ++t) {
#pragma unroll
      for (int r = 0; r < 8; ++r) {
        const float p = __expf(sacc[t][r] - mnew[r]);
        psum[r] += p;
        Pw[(r + khalf * 8) * 72 + t * 16 + lane16] = (bf16_t)p;  // C->LDS
      }
    }
#pragma unroll
    for (int r = 0; r < 8; ++r) {
      float ps = psum[r];
      for (int off = 1; off < 16; off <<= 1) ps += __shfl_xor(ps, off, 32);
      lsum[r] = lsum[r] * alpha[r] + ps;
    }
#pragma unroll
    for (int t = 0; t < 4; ++t)
#pragma unroll
      for (int r = 0; r < 8; ++r) oacc[t][r] *= alpha[r];

    __syncthreads();  // P tile visible (and prior reads done)

    // ---- O += P @ V ----
#pragma unroll
    for (int s = 0; s < 2; ++s) {
      V16 pf;  // re-read P in A-layout
      const bf16_t* prow = Pw + lane16 * 72 + s * 32 + khalf * 8;
      pf.h[0] = *(const v8bf*)prow;
      pf.h[1] = *(const v8bf*)(prow + 16);
#pragma unroll
      for (int t = 0; t < 4; ++t) {
        const int d = t * 16 + lane16;
        const bf16_t* vrow = Vbh + (size_t)d * L_ + kb + s * 32 + khalf * 8;
        V16 vf;
        vf.h[0] = *(const v8bf*)vrow;
        vf.h[1] = *(const v8bf*)(vrow + 16);
        oacc[t] = wmma_bf16f32(pf.v, vf.v, oacc[t]);
      }
    }
    __syncthreads();
  }

  // ---- normalize + write [tok][h*64+d] bf16 for the out-proj GEMM ----
#pragma unroll
  for (int t = 0; t < 4; ++t) {
#pragma unroll
    for (int r = 0; r < 8; ++r) {
      const int ql = qr0 + r + khalf * 8;
      const float o = oacc[t][r] / lsum[r];
      attnout[((size_t)(b * L_ + ql)) * DIM_ + h * HD_ + t * 16 + lane16] =
          (bf16_t)o;
    }
  }
}

// ---------------------------------------------------------------------------
// Output projection: attnout(8192x512 bf16) @ wproj(512x512 bf16) -> fp32 out.
// ---------------------------------------------------------------------------
__global__ void gemm_proj_kernel(const bf16_t* __restrict__ attnb,
                                 const bf16_t* __restrict__ wprojb,
                                 float* __restrict__ out) {
  __shared__ bf16_t As[256 * 40];
  __shared__ bf16_t Bs[64 * 48];
  const int rowBase = blockIdx.y * 256;
  const int colBase = blockIdx.x * 64;
  v8f acc[2][4];
  gemm_tile(attnb, DIM_, wprojb, DIM_, DIM_, rowBase, colBase, As, Bs, acc);

  const int tid = threadIdx.x;
  const int w = tid >> 5, lane = tid & 31;
  const int lane16 = lane & 15, khalf = lane >> 4;
#pragma unroll
  for (int u = 0; u < 2; ++u)
#pragma unroll
    for (int t = 0; t < 4; ++t)
#pragma unroll
      for (int r = 0; r < 8; ++r)
        out[(size_t)(rowBase + w * 32 + u * 16 + r + khalf * 8) * DIM_ +
            colBase + t * 16 + lane16] = acc[u][t][r];
}

// ---------------------------------------------------------------------------
// Workspace layout (bytes), total 44,040,192 (42 MB)
// ---------------------------------------------------------------------------
#define WS_XB     ((size_t)0)
#define WS_WQKV   ((size_t)8388608)
#define WS_WPROJ  ((size_t)9961472)
#define WS_Q      ((size_t)10485760)
#define WS_K      ((size_t)18874368)
#define WS_VT     ((size_t)27262976)
#define WS_ATT    ((size_t)35651584)

extern "C" void kernel_launch(void* const* d_in, const int* in_sizes, int n_in,
                              void* d_out, int out_size, void* d_ws,
                              size_t ws_size, hipStream_t stream) {
  const float* x = (const float*)d_in[0];
  const unsigned char* mask = (const unsigned char*)d_in[1];
  const float* w_qkv = (const float*)d_in[2];
  const float* w_proj = (const float*)d_in[3];
  float* out = (float*)d_out;
  char* ws = (char*)d_ws;

  bf16_t* xb = (bf16_t*)(ws + WS_XB);
  bf16_t* wqkvb = (bf16_t*)(ws + WS_WQKV);
  bf16_t* wprojb = (bf16_t*)(ws + WS_WPROJ);
  bf16_t* Qb = (bf16_t*)(ws + WS_Q);
  bf16_t* Kb = (bf16_t*)(ws + WS_K);
  bf16_t* Vtb = (bf16_t*)(ws + WS_VT);
  bf16_t* attnb = (bf16_t*)(ws + WS_ATT);

  const int nx = NTOK_ * DIM_;       // 4,194,304
  const int nwq = DIM_ * N_QKV_;     //   786,432
  const int nwp = DIM_ * DIM_;       //   262,144
  f32_to_bf16_kernel<<<(nx + 255) / 256, 256, 0, stream>>>(x, xb, nx);
  f32_to_bf16_kernel<<<(nwq + 255) / 256, 256, 0, stream>>>(w_qkv, wqkvb, nwq);
  f32_to_bf16_kernel<<<(nwp + 255) / 256, 256, 0, stream>>>(w_proj, wprojb, nwp);

  dim3 g1(N_QKV_ / 64, NTOK_ / 256);  // (24, 32)
  gemm_qkv_rope_kernel<<<g1, 256, 0, stream>>>(xb, wqkvb, Qb, Kb, Vtb);

  flash_attn_kernel<<<B_ * NH_ * (L_ / 64), 128, 0, stream>>>(Qb, Kb, Vtb,
                                                              mask, attnb);

  dim3 g4(DIM_ / 64, NTOK_ / 256);  // (8, 32)
  gemm_proj_kernel<<<g4, 256, 0, stream>>>(attnb, wprojb, out);
}
